// MultiModalAttention_15023795601609
// MI455X (gfx1250) — matrix operand order repaired and checked by reading, not compiled, not run on hardware
//
#include <hip/hip_runtime.h>

typedef __attribute__((ext_vector_type(16))) _Float16 v16h;
typedef __attribute__((ext_vector_type(8)))  _Float16 v8h;
typedef __attribute__((ext_vector_type(8)))  float    v8f;

union Frag { v16h v; v8h h[2]; };

#define BB     2
#define SS     1024
#define NVV    256
#define DDIM   4096
#define VDIM   768
#define NH     32
#define NKV    4
#define HD     128
#define KVLEN  1280
#define KVP    1312        // padded kv length (multiple of 32)
#define NEGV   (-1e30f)

__device__ __forceinline__ v8f wmma16(const Frag& a, const Frag& b, v8f c) {
    return __builtin_amdgcn_wmma_f32_16x16x32_f16(
        /*neg_a=*/false, a.v, /*neg_b=*/false, b.v,
        /*c_mod=*/(short)0, c, /*reuse_a=*/false, /*reuse_b=*/false);
}

// ---------------------------------------------------------------- convert
__global__ __launch_bounds__(256)
void f32_to_f16_kernel(const float* __restrict__ s, _Float16* __restrict__ d, int n) {
    int i = blockIdx.x * 256 + threadIdx.x;
    if (i < n) d[i] = (_Float16)s[i];
}

// ---------------------------------------------------------------- GEMM: C = A(MxK) @ W(NxK)^T
// block = 128 threads (4 waves); block tile 64x128; wave tile 32x64 (2x4 wmma)
__global__ __launch_bounds__(128)
void gemm_wmma(const _Float16* __restrict__ A, const _Float16* __restrict__ Bw,
               float* __restrict__ Cf, _Float16* __restrict__ Ch,
               int M, int N, int K)
{
    const int lane = threadIdx.x & 31;
    const int wave = threadIdx.x >> 5;
    const int l15  = lane & 15;
    const int hi   = lane >> 4;
    const int m0 = blockIdx.y * 64  + (wave >> 1) * 32;
    const int n0 = blockIdx.x * 128 + (wave & 1) * 64;

    const _Float16* a0p = A + (size_t)(m0      + l15) * K;
    const _Float16* a1p = A + (size_t)(m0 + 16 + l15) * K;
    const _Float16* bp0 = Bw + (size_t)(n0      + l15) * K;
    const _Float16* bp1 = Bw + (size_t)(n0 + 16 + l15) * K;
    const _Float16* bp2 = Bw + (size_t)(n0 + 32 + l15) * K;
    const _Float16* bp3 = Bw + (size_t)(n0 + 48 + l15) * K;

    v8f acc[2][4] = {};

    for (int k0 = 0; k0 < K; k0 += 32) {
        Frag a0, a1, b0, b1, b2, b3;
        // A fragment: lane row = m, K = hi*16 + [0..16) contiguous
        const _Float16* pa0 = a0p + k0 + hi * 16;
        a0.h[0] = *(const v8h*)(pa0);
        a0.h[1] = *(const v8h*)(pa0 + 8);
        const _Float16* pa1 = a1p + k0 + hi * 16;
        a1.h[0] = *(const v8h*)(pa1);
        a1.h[1] = *(const v8h*)(pa1 + 8);
        // B fragments: lane n holds W row n, K = hi*16 + [0..16) contiguous
        const _Float16* pb0 = bp0 + k0 + hi * 16;
        b0.h[0] = *(const v8h*)(pb0);
        b0.h[1] = *(const v8h*)(pb0 + 8);
        const _Float16* pb1 = bp1 + k0 + hi * 16;
        b1.h[0] = *(const v8h*)(pb1);
        b1.h[1] = *(const v8h*)(pb1 + 8);
        const _Float16* pb2 = bp2 + k0 + hi * 16;
        b2.h[0] = *(const v8h*)(pb2);
        b2.h[1] = *(const v8h*)(pb2 + 8);
        const _Float16* pb3 = bp3 + k0 + hi * 16;
        b3.h[0] = *(const v8h*)(pb3);
        b3.h[1] = *(const v8h*)(pb3 + 8);

        if (k0 + 32 < K) {                       // global_prefetch_b8 next K tile
            __builtin_prefetch(pa0 + 32, 0, 1);
            __builtin_prefetch(pb0 + 32, 0, 1);
        }

        acc[0][0] = wmma16(a0, b0, acc[0][0]);
        acc[0][1] = wmma16(a0, b1, acc[0][1]);
        acc[0][2] = wmma16(a0, b2, acc[0][2]);
        acc[0][3] = wmma16(a0, b3, acc[0][3]);
        acc[1][0] = wmma16(a1, b0, acc[1][0]);
        acc[1][1] = wmma16(a1, b1, acc[1][1]);
        acc[1][2] = wmma16(a1, b2, acc[1][2]);
        acc[1][3] = wmma16(a1, b3, acc[1][3]);
    }

    #pragma unroll
    for (int i = 0; i < 2; ++i) {
        #pragma unroll
        for (int j = 0; j < 4; ++j) {
            #pragma unroll
            for (int r = 0; r < 8; ++r) {
                const int row = m0 + i * 16 + r + hi * 8;  // D layout: VGPR r -> M=r / r+8
                const int col = n0 + j * 16 + l15;
                if (Ch) Ch[(size_t)row * N + col] = (_Float16)acc[i][j][r];
                else    Cf[(size_t)row * N + col] = acc[i][j][r];
            }
        }
    }
}

// ---------------------------------------------------------------- RoPE on Q: qraw[b,s,h,d] -> Qh[b,h,s,d]
__global__ __launch_bounds__(256)
void rope_q_kernel(const _Float16* __restrict__ qraw, const float* __restrict__ cs,
                   const float* __restrict__ sn, _Float16* __restrict__ Qh)
{
    const int t = blockIdx.x * 256 + threadIdx.x;        // B*H*S*HD = 8388608 exact
    const int d = t & 127;
    const int s = (t >> 7) & 1023;
    const int h = (t >> 17) & 31;
    const int b = t >> 22;
    const size_t src = ((size_t)(b * SS + s)) * DDIM + h * HD;
    const float q  = (float)qraw[src + d];
    const float qp = (d < 64) ? -(float)qraw[src + d + 64] : (float)qraw[src + d - 64];
    const float c = cs[s * HD + d], si = sn[s * HD + d];
    Qh[(((size_t)(b * NH + h)) * SS + s) * HD + d] = (_Float16)(q * c + qp * si);
}

// ------------------------------------------- build K: concat(vision-k, text-k) + RoPE, pad to KVP
__global__ __launch_bounds__(256)
void build_k_kernel(const _Float16* __restrict__ vkvraw, const _Float16* __restrict__ kraw,
                    const float* __restrict__ cs, const float* __restrict__ sn,
                    _Float16* __restrict__ Kh)
{
    const int t = blockIdx.x * 256 + threadIdx.x;        // B*NKV*KVP*HD = 1343488 exact
    const int d  = t & 127;
    int rest = t >> 7;
    const int tt = rest % KVP; rest /= KVP;
    const int g  = rest & 3;
    const int b  = rest >> 2;
    const size_t dst = (((size_t)(b * NKV + g)) * KVP + tt) * HD + d;
    if (tt >= KVLEN) { Kh[dst] = (_Float16)0.0f; return; }
    const int dp = (d < 64) ? d + 64 : d - 64;
    float q, qp;
    if (tt < NVV) {   // vision k = vkvraw[:, :512]
        const size_t row = ((size_t)(b * NVV + tt)) * (2 * NKV * HD);
        q  = (float)vkvraw[row + g * HD + d];
        qp = (float)vkvraw[row + g * HD + dp];
    } else {          // text k
        const size_t row = ((size_t)(b * SS + (tt - NVV))) * (NKV * HD);
        q  = (float)kraw[row + g * HD + d];
        qp = (float)kraw[row + g * HD + dp];
    }
    if (d < 64) qp = -qp;
    const float c = cs[tt * HD + d], si = sn[tt * HD + d];
    Kh[dst] = (_Float16)(q * c + qp * si);
}

// ------------------------------------------- build V^T: concat(vision-v, text-v), transpose, pad
__global__ __launch_bounds__(256)
void build_vt_kernel(const _Float16* __restrict__ vkvraw, const _Float16* __restrict__ vraw,
                     _Float16* __restrict__ VTh)
{
    const int t = blockIdx.x * 256 + threadIdx.x;
    const int d  = t & 127;
    int rest = t >> 7;
    const int tt = rest % KVP; rest /= KVP;
    const int g  = rest & 3;
    const int b  = rest >> 2;
    const size_t dst = (((size_t)(b * NKV + g)) * HD + d) * KVP + tt;
    if (tt >= KVLEN) { VTh[dst] = (_Float16)0.0f; return; }
    _Float16 v;
    if (tt < NVV)
        v = vkvraw[((size_t)(b * NVV + tt)) * (2 * NKV * HD) + NKV * HD + g * HD + d];
    else
        v = vraw[((size_t)(b * SS + (tt - NVV))) * (NKV * HD) + g * HD + d];
    VTh[dst] = v;
}

// ---------------------------------------------------------------- flash attention
// 8 waves/block; each wave owns one (b, h, 16-row q tile); online softmax over 32-col kv tiles
__global__ __launch_bounds__(256)
void flash_attn_kernel(const _Float16* __restrict__ Qh, const _Float16* __restrict__ Kh,
                       const _Float16* __restrict__ VTh, _Float16* __restrict__ Oh)
{
    __shared__ _Float16 lds[8 * 16 * 32];     // per-wave private 16x32 P tile
    const int lane = threadIdx.x & 31;
    const int wv   = threadIdx.x >> 5;
    const int l15  = lane & 15;
    const int hi   = lane >> 4;
    const int wid  = blockIdx.x * 8 + wv;     // 4096 wave-tiles total
    const int i0 = (wid & 63) * 16;
    const int h  = (wid >> 6) & 31;
    const int b  = wid >> 11;
    const int g  = h >> 3;                    // REP = 8

    const _Float16* Qp = Qh  + (((size_t)(b * NH + h)) * SS + i0 + l15) * HD;
    const _Float16* Kp = Kh  + ((size_t)(b * NKV + g)) * KVP * HD;
    const _Float16* Vp = VTh + ((size_t)(b * NKV + g)) * HD * KVP;
    _Float16* P = lds + wv * 512;

    Frag qf[4];
    #pragma unroll
    for (int kk = 0; kk < 4; ++kk) {
        const _Float16* p = Qp + kk * 32 + hi * 16;
        qf[kk].h[0] = *(const v8h*)p;
        qf[kk].h[1] = *(const v8h*)(p + 8);
    }

    v8f O[8] = {};
    float mr[8], lr[8], al[8];
    const float NEG_INF = -__builtin_inff();
    #pragma unroll
    for (int r = 0; r < 8; ++r) { mr[r] = NEG_INF; lr[r] = 0.0f; }

    const float scale = 0.0883883476483184f;  // 128^-0.5
    const int jend = i0 + 272;                // max allowed col + 1 (causal offset 256)

    for (int t0 = 0; t0 < jend; t0 += 32) {
        v8f s0 = {}, s1 = {};
        const _Float16* k0p = Kp + (size_t)(t0      + l15) * HD + hi * 16;
        const _Float16* k1p = Kp + (size_t)(t0 + 16 + l15) * HD + hi * 16;
        #pragma unroll
        for (int kk = 0; kk < 4; ++kk) {
            Frag kb0, kb1;
            kb0.h[0] = *(const v8h*)(k0p + kk * 32);
            kb0.h[1] = *(const v8h*)(k0p + kk * 32 + 8);
            kb1.h[0] = *(const v8h*)(k1p + kk * 32);
            kb1.h[1] = *(const v8h*)(k1p + kk * 32 + 8);
            s0 = wmma16(qf[kk], kb0, s0);
            s1 = wmma16(qf[kk], kb1, s1);
        }
        #pragma unroll
        for (int r = 0; r < 8; ++r) {
            const int irow = i0 + r + hi * 8;
            float v0 = s0[r] * scale; if (t0      + l15 > irow + 256) v0 = NEGV;
            float v1 = s1[r] * scale; if (t0 + 16 + l15 > irow + 256) v1 = NEGV;
            float mt = fmaxf(v0, v1);            // row reduce across the 16 N-lanes
            mt = fmaxf(mt, __shfl_xor(mt, 1, 32));
            mt = fmaxf(mt, __shfl_xor(mt, 2, 32));
            mt = fmaxf(mt, __shfl_xor(mt, 4, 32));
            mt = fmaxf(mt, __shfl_xor(mt, 8, 32));
            const float mn = fmaxf(mr[r], mt);
            const float a  = __expf(mr[r] - mn);
            const float p0 = __expf(v0 - mn);
            const float p1 = __expf(v1 - mn);
            float rs = p0 + p1;
            rs += __shfl_xor(rs, 1, 32);
            rs += __shfl_xor(rs, 2, 32);
            rs += __shfl_xor(rs, 4, 32);
            rs += __shfl_xor(rs, 8, 32);
            lr[r] = lr[r] * a + rs;
            mr[r] = mn;
            al[r] = a;
            const int prow = r + hi * 8;         // D layout -> LDS row-major P
            P[prow * 32 + l15]      = (_Float16)p0;
            P[prow * 32 + 16 + l15] = (_Float16)p1;
        }
        #pragma unroll
        for (int c = 0; c < 8; ++c)
            #pragma unroll
            for (int r = 0; r < 8; ++r) O[c][r] *= al[r];

        Frag pf;   // reload P in A-fragment layout (ds_load, in-order per wave)
        {
            const _Float16* pp = P + l15 * 32 + hi * 16;
            pf.h[0] = *(const v8h*)pp;
            pf.h[1] = *(const v8h*)(pp + 8);
        }
        #pragma unroll
        for (int c = 0; c < 8; ++c) {
            Frag vf;   // B fragment from V^T rows: contiguous in t
            const _Float16* vp = Vp + (size_t)(c * 16 + l15) * KVP + t0 + hi * 16;
            vf.h[0] = *(const v8h*)vp;
            vf.h[1] = *(const v8h*)(vp + 8);
            O[c] = wmma16(pf, vf, O[c]);
        }
    }

    #pragma unroll
    for (int r = 0; r < 8; ++r) {
        const float inv = 1.0f / lr[r];
        const int row = i0 + r + hi * 8;
        _Float16* op = Oh + ((size_t)(b * SS + row)) * (NH * HD) + h * HD;
        #pragma unroll
        for (int c = 0; c < 8; ++c)
            op[c * 16 + l15] = (_Float16)(O[c][r] * inv);
    }
}

// ---------------------------------------------------------------- host side
extern "C" void kernel_launch(void* const* d_in, const int* in_sizes, int n_in,
                              void* d_out, int out_size, void* d_ws, size_t ws_size,
                              hipStream_t stream)
{
    (void)in_sizes; (void)n_in; (void)out_size; (void)ws_size;
    const float* x    = (const float*)d_in[0];
    const float* vfea = (const float*)d_in[1];
    const float* wq   = (const float*)d_in[2];
    const float* wk   = (const float*)d_in[3];
    const float* wv   = (const float*)d_in[4];
    const float* wo   = (const float*)d_in[5];
    const float* wkv  = (const float*)d_in[6];
    const float* cs   = (const float*)d_in[7];
    const float* sn   = (const float*)d_in[8];
    float* out = (float*)d_out;

    char* ws = (char*)d_ws;
    size_t off = 0;
    auto take = [&](size_t bytes) -> void* {
        void* p = ws + off;
        off += (bytes + 255) & ~(size_t)255;
        return p;
    };
    _Float16* xh     = (_Float16*)take((size_t)BB * SS * DDIM * 2);          // 16 MB
    _Float16* vfh    = (_Float16*)take((size_t)BB * NVV * VDIM * 2);
    _Float16* wqh    = (_Float16*)take((size_t)DDIM * DDIM * 2);             // 32 MB
    _Float16* wkh    = (_Float16*)take((size_t)NKV * HD * DDIM * 2);
    _Float16* wvh    = (_Float16*)take((size_t)NKV * HD * DDIM * 2);
    _Float16* woh    = (_Float16*)take((size_t)DDIM * DDIM * 2);             // 32 MB
    _Float16* wkvh   = (_Float16*)take((size_t)2 * NKV * HD * VDIM * 2);
    _Float16* qraw   = (_Float16*)take((size_t)BB * SS * DDIM * 2);          // reused as attn_out
    _Float16* kraw   = (_Float16*)take((size_t)BB * SS * NKV * HD * 2);
    _Float16* vraw   = (_Float16*)take((size_t)BB * SS * NKV * HD * 2);
    _Float16* vkvraw = (_Float16*)take((size_t)BB * NVV * 2 * NKV * HD * 2);
    _Float16* Qh     = (_Float16*)take((size_t)BB * NH * SS * HD * 2);
    _Float16* Kh     = (_Float16*)take((size_t)BB * NKV * KVP * HD * 2);
    _Float16* VTh    = (_Float16*)take((size_t)BB * NKV * HD * KVP * 2);
    _Float16* attn   = qraw;   // alias: qraw is dead after rope_q

    auto cvt = [&](const float* s, _Float16* d, int n) {
        f32_to_f16_kernel<<<(n + 255) / 256, 256, 0, stream>>>(s, d, n);
    };
    cvt(x,    xh,   BB * SS * DDIM);
    cvt(vfea, vfh,  BB * NVV * VDIM);
    cvt(wq,   wqh,  DDIM * DDIM);
    cvt(wk,   wkh,  NKV * HD * DDIM);
    cvt(wv,   wvh,  NKV * HD * DDIM);
    cvt(wo,   woh,  DDIM * DDIM);
    cvt(wkv,  wkvh, 2 * NKV * HD * VDIM);

    auto gemm = [&](const _Float16* A, const _Float16* Bw, float* Cf, _Float16* Ch,
                    int M, int N, int K) {
        dim3 g(N / 128, M / 64);
        gemm_wmma<<<g, 128, 0, stream>>>(A, Bw, Cf, Ch, M, N, K);
    };
    gemm(xh,  wqh,  nullptr, qraw,   BB * SS,  NH * HD,       DDIM);   // Q proj
    gemm(xh,  wkh,  nullptr, kraw,   BB * SS,  NKV * HD,      DDIM);   // K proj
    gemm(xh,  wvh,  nullptr, vraw,   BB * SS,  NKV * HD,      DDIM);   // V proj
    gemm(vfh, wkvh, nullptr, vkvraw, BB * NVV, 2 * NKV * HD,  VDIM);   // vision KV proj

    rope_q_kernel<<<(BB * NH * SS * HD) / 256, 256, 0, stream>>>(qraw, cs, sn, Qh);
    build_k_kernel<<<(BB * NKV * KVP * HD) / 256, 256, 0, stream>>>(vkvraw, kraw, cs, sn, Kh);
    build_vt_kernel<<<(BB * NKV * KVP * HD) / 256, 256, 0, stream>>>(vkvraw, vraw, VTh);

    flash_attn_kernel<<<(BB * NH * (SS / 16)) / 8, 256, 0, stream>>>(Qh, Kh, VTh, attn);

    gemm(attn, woh, out, nullptr, BB * SS, DDIM, DDIM);                // output proj
}
